// NGP_15968688406828
// MI455X (gfx1250) — compile-verified, matrix operand order checked by reading, not run on hardware
//
#include <hip/hip_runtime.h>
#include <hip/hip_bf16.h>

typedef __attribute__((ext_vector_type(16))) _Float16 v16h;
typedef __attribute__((ext_vector_type(8)))  _Float16 v8h;
typedef __attribute__((ext_vector_type(8)))  float    v8f;

#define RES   16
#define NFEAT 8
#define DH    128
#define DIN   11
#define W0S   30.0f
#define WPB   4   // waves per block

union V16U { v16h v; v8h h[2]; };

static __device__ __forceinline__ v8f wmma16(v16h a, v16h b, v8f c) {
  // D = A(16x32 f16) * B(32x16 f16) + C(16x16 f32)
  return __builtin_amdgcn_wmma_f32_16x16x32_f16(false, a, false, b, (short)0, c, false, false);
}

// A-fragment from a row-major [16][stride] f16 LDS tile.
// Lane l: row m = l&15, K-base kb = (l>>4)*8; element e -> K = kb + (e<8 ? e : e+8).
static __device__ __forceinline__ v16h frag_rowmajor(const _Float16* buf, int stride, int k0, int lane) {
  int m  = lane & 15;
  int kb = (lane >> 4) * 8;
  const _Float16* p = buf + m * stride + k0 + kb;
  V16U u;
  u.h[0] = *(const v8h*)(p);
  u.h[1] = *(const v8h*)(p + 16);
  return u.v;
}

// B-fragment pre-packed as [frag][lane][16] f16 (contiguous per lane).
static __device__ __forceinline__ v16h frag_packed(const _Float16* buf, int frag, int lane) {
  const _Float16* p = buf + (frag * 32 + lane) * 16;
  V16U u;
  u.h[0] = *(const v8h*)(p);
  u.h[1] = *(const v8h*)(p + 8);
  return u.v;
}

__global__ __launch_bounds__(32 * WPB) void ngp_sdf_fused(
    const float* __restrict__ pos, const float* __restrict__ grid,
    const float* __restrict__ W0, const float* __restrict__ b0,
    const float* __restrict__ W1, const float* __restrict__ b1,
    const float* __restrict__ W2, const float* __restrict__ b2,
    float* __restrict__ outS, float* __restrict__ outG, float* __restrict__ outM,
    int ntiles)
{
  // Weight fragments, packed in WMMA B layout (shared by all waves in block)
  __shared__ __align__(16) _Float16 sW1T[4 * 8 * 32 * 16]; // forward : B[k][n] = W1[n][k]
  __shared__ __align__(16) _Float16 sW1B[4 * 8 * 32 * 16]; // backward: B[k][n] = W1[k][n]
  __shared__ __align__(16) _Float16 sW0F[8 * 32 * 16];     // forward : 32(K pad) x 128
  __shared__ __align__(16) _Float16 sW0B[4 * 32 * 16];     // backward: 128 x 16(N pad)
  // Per-wave staging
  __shared__ __align__(16) _Float16 sXH[WPB][16 * 128];    // x_pad then h0, row-major [m][k]
  __shared__ __align__(16) _Float16 sC0[WPB][128 * 16];    // 30*cos(z0), layout [n][m]
  __shared__ __align__(16) _Float16 sG [WPB][16 * 128];    // g1 then g0, row-major [m][k]
  __shared__ float sGX[WPB][16 * 16];                      // gx = W0^T g0
  __shared__ float sDF[WPB][16 * 24];                      // d feat_j / d pos_k  (x15)

  const int tid = threadIdx.x;

  // ---- pack weights into fragment order ----
  for (int t = tid; t < 4 * 8 * 32 * 16; t += blockDim.x) {
    int e = t & 15, lane = (t >> 4) & 31, nt = (t >> 9) & 7, kc = t >> 12;
    int n = nt * 16 + (lane & 15);
    int K = kc * 32 + ((lane >> 4) * 8) + (e < 8 ? e : e + 8);
    sW1T[t] = (_Float16)W1[n * DH + K];   // z1[n] = sum_k h0[k] * W1[n][k]
    sW1B[t] = (_Float16)W1[K * DH + n];   // t0[n] = sum_k g1[k] * W1[k][n]
  }
  for (int t = tid; t < 8 * 32 * 16; t += blockDim.x) {
    int e = t & 15, lane = (t >> 4) & 31, nt = t >> 9;
    int n = nt * 16 + (lane & 15);
    int K = ((lane >> 4) * 8) + (e < 8 ? e : e + 8);
    sW0F[t] = (K < DIN) ? (_Float16)W0[n * DIN + K] : (_Float16)0.0f;
  }
  for (int t = tid; t < 4 * 32 * 16; t += blockDim.x) {
    int e = t & 15, lane = (t >> 4) & 31, kc = t >> 9;
    int n = lane & 15;
    int K = kc * 32 + ((lane >> 4) * 8) + (e < 8 ? e : e + 8);
    sW0B[t] = (n < DIN) ? (_Float16)W0[K * DIN + n] : (_Float16)0.0f;
  }
  __syncthreads();

  const int wid  = tid >> 5;
  const int lane = tid & 31;
  _Float16* xh  = sXH[wid];
  _Float16* c0p = sC0[wid];
  _Float16* gls = sG[wid];
  float*    gxs = sGX[wid];
  float*    dfp = sDF[wid];
  const int ncol  = lane & 15;
  const int mbase = (lane >> 4) * 8;
  const int gwave = blockIdx.x * WPB + wid;
  const int nwave = gridDim.x * WPB;

  for (int tile = gwave; tile < ntiles; tile += nwave) {
    const int base = tile * 16;

    // ---- phase 1: trilinear encode, build x_pad (16x32, zero padded) ----
    {
      v8h z = {};
      ((v8h*)xh)[lane]      = z;   // zero 16x32 halves cooperatively
      ((v8h*)xh)[lane + 32] = z;
    }
    if (lane < 16) {
      const int pidx = base + lane;
      float p3[3];
      p3[0] = pos[pidx * 3 + 0];
      p3[1] = pos[pidx * 3 + 1];
      p3[2] = pos[pidx * 3 + 2];
      float fr[3]; int i0[3];
      #pragma unroll
      for (int k = 0; k < 3; k++) {
        float q  = fminf(fmaxf(p3[k], 0.f), 1.f) * (float)(RES - 1);
        float fl = fminf(floorf(q), (float)(RES - 2));
        fr[k] = q - fl;
        i0[k] = (int)fl;
      }
      float fe[8], d0[8], d1[8], d2[8];
      #pragma unroll
      for (int j = 0; j < 8; j++) { fe[j] = 0.f; d0[j] = 0.f; d1[j] = 0.f; d2[j] = 0.f; }
      #pragma unroll
      for (int c = 0; c < 8; c++) {
        int dx = (c >> 2) & 1, dy = (c >> 1) & 1, dz = c & 1;
        float wx = dx ? fr[0] : 1.f - fr[0];
        float wy = dy ? fr[1] : 1.f - fr[1];
        float wz = dz ? fr[2] : 1.f - fr[2];
        float sx = dx ? 1.f : -1.f, sy = dy ? 1.f : -1.f, sz = dz ? 1.f : -1.f;
        const float* gp = grid + ((((i0[0] + dx) * RES + (i0[1] + dy)) * RES + (i0[2] + dz)) * NFEAT);
        float w   = wx * wy * wz;
        float wdx = sx * wy * wz, wdy = wx * sy * wz, wdz = wx * wy * sz;
        #pragma unroll
        for (int j = 0; j < 8; j++) {
          float v = gp[j];
          fe[j] += w * v; d0[j] += wdx * v; d1[j] += wdy * v; d2[j] += wdz * v;
        }
      }
      #pragma unroll
      for (int j = 0; j < 8; j++) {
        dfp[lane * 24 + j]      = d0[j] * (float)(RES - 1);
        dfp[lane * 24 + 8 + j]  = d1[j] * (float)(RES - 1);
        dfp[lane * 24 + 16 + j] = d2[j] * (float)(RES - 1);
        xh[lane * 32 + j] = (_Float16)fe[j];
      }
      xh[lane * 32 + 8]  = (_Float16)p3[0];
      xh[lane * 32 + 9]  = (_Float16)p3[1];
      xh[lane * 32 + 10] = (_Float16)p3[2];
    }

    // ---- phase 2: layer0  z0 = x @ W0^T  (16x32 * 32x128) ----
    v16h a0 = frag_rowmajor(xh, 32, 0, lane);
    v8f acc[8];
    #pragma unroll
    for (int nt = 0; nt < 8; nt++) {
      v8f cz = {};
      acc[nt] = wmma16(a0, frag_packed(sW0F, nt, lane), cz);
    }
    // siren act: h0 = sin(30(z0+b0)), stash 30*cos factor
    #pragma unroll
    for (int nt = 0; nt < 8; nt++) {
      int n = nt * 16 + ncol;
      float bb = W0S * b0[n];
      v8h cv;
      #pragma unroll
      for (int r = 0; r < 8; r++) {
        float arg = W0S * acc[nt][r] + bb;
        xh[(mbase + r) * 128 + n] = (_Float16)__sinf(arg);
        cv[r] = (_Float16)(W0S * __cosf(arg));
      }
      *(v8h*)(c0p + n * 16 + mbase) = cv;   // [n][m] layout, contiguous in m
    }

    // ---- phase 3: layer1 forward  z1 = h0 @ W1^T  (16x128 * 128x128) ----
    v16h af[4];
    #pragma unroll
    for (int kc = 0; kc < 4; kc++) af[kc] = frag_rowmajor(xh, 128, kc * 32, lane);
    #pragma unroll
    for (int nt = 0; nt < 8; nt++) {
      v8f cz = {};
      #pragma unroll
      for (int kc = 0; kc < 4; kc++)
        cz = wmma16(af[kc], frag_packed(sW1T, kc * 8 + nt, lane), cz);
      acc[nt] = cz;
    }

    // ---- phase 4: h1, sdf, g1 = W2 * 30cos(z1) ----
    float part[8];
    #pragma unroll
    for (int r = 0; r < 8; r++) part[r] = 0.f;
    #pragma unroll
    for (int nt = 0; nt < 8; nt++) {
      int n = nt * 16 + ncol;
      float bb  = W0S * b1[n];
      float w2n = W2[n];
      #pragma unroll
      for (int r = 0; r < 8; r++) {
        float arg = W0S * acc[nt][r] + bb;
        part[r] += __sinf(arg) * w2n;
        gls[(mbase + r) * 128 + n] = (_Float16)(w2n * W0S * __cosf(arg));
      }
    }
    #pragma unroll
    for (int d = 1; d < 16; d <<= 1) {
      #pragma unroll
      for (int r = 0; r < 8; r++) part[r] += __shfl_xor(part[r], d, 32);
    }
    {
      float bb2 = b2[0];
      if (lane == 0) {
        #pragma unroll
        for (int r = 0; r < 8; r++) outS[base + r] = part[r] + bb2;
      }
      if (lane == 16) {
        #pragma unroll
        for (int r = 0; r < 8; r++) outS[base + 8 + r] = part[r] + bb2;
      }
    }

    // ---- phase 5: backward layer1  g0 = (g1 @ W1) * 30cos(z0) ----
    #pragma unroll
    for (int kc = 0; kc < 4; kc++) af[kc] = frag_rowmajor(gls, 128, kc * 32, lane);
    #pragma unroll
    for (int nt = 0; nt < 8; nt++) {
      v8f cz = {};
      #pragma unroll
      for (int kc = 0; kc < 4; kc++)
        cz = wmma16(af[kc], frag_packed(sW1B, kc * 8 + nt, lane), cz);
      int n = nt * 16 + ncol;
      v8h c0v = *(const v8h*)(c0p + n * 16 + mbase);
      #pragma unroll
      for (int r = 0; r < 8; r++)
        gls[(mbase + r) * 128 + n] = (_Float16)(cz[r] * (float)c0v[r]);
    }

    // ---- phase 6: backward layer0  gx = g0 @ W0  (16x128 * 128x16) ----
    #pragma unroll
    for (int kc = 0; kc < 4; kc++) af[kc] = frag_rowmajor(gls, 128, kc * 32, lane);
    v8f gacc = {};
    #pragma unroll
    for (int kc = 0; kc < 4; kc++)
      gacc = wmma16(af[kc], frag_packed(sW0B, kc, lane), gacc);
    #pragma unroll
    for (int r = 0; r < 8; r++)
      gxs[(mbase + r) * 16 + ncol] = gacc[r];

    // ---- phase 7: combine with trilinear derivative ----
    if (lane < 16) {
      int m = lane, pidx = base + m;
      #pragma unroll
      for (int k = 0; k < 3; k++) {
        float s = gxs[m * 16 + 8 + k];        // direct pos term
        #pragma unroll
        for (int j = 0; j < 8; j++)
          s += gxs[m * 16 + j] * dfp[m * 24 + k * 8 + j];
        outG[pidx * 3 + k] = s;
      }
      outM[pidx] = 1.0f;
    }
  }
}

extern "C" void kernel_launch(void* const* d_in, const int* in_sizes, int n_in,
                              void* d_out, int out_size, void* d_ws, size_t ws_size,
                              hipStream_t stream) {
  const float* pos  = (const float*)d_in[0];
  const float* grid = (const float*)d_in[1];
  const float* W0   = (const float*)d_in[2];
  const float* b0   = (const float*)d_in[3];
  const float* W1   = (const float*)d_in[4];
  const float* b1   = (const float*)d_in[5];
  const float* W2   = (const float*)d_in[6];
  const float* b2   = (const float*)d_in[7];
  int N = in_sizes[0] / 3;
  int ntiles = N / 16;
  float* outS = (float*)d_out;        // sdf  [N]
  float* outG = outS + N;             // grad [N,3]
  float* outM = outS + 4 * (size_t)N; // mask [N]
  int blocks = ntiles / (WPB * 4);
  if (blocks < 1) blocks = 1;
  if (blocks > 2048) blocks = 2048;
  ngp_sdf_fused<<<blocks, 32 * WPB, 0, stream>>>(pos, grid, W0, b0, W1, b1, W2, b2,
                                                 outS, outG, outM, ntiles);
}
